// UnrolledLM_63788854280361
// MI455X (gfx1250) — matrix operand order repaired and checked by reading, not compiled, not run on hardware
//
#include <hip/hip_runtime.h>
#include <hip/hip_bf16.h>
#include <stdint.h>

// Problem constants (match reference)
#define H_DIM   1024
#define IN_DIM  512
#define OUT_DIM 8192
#define T_STEPS 128
#define B_DIM   32
#define KCAT    3072   // (K-1)*H
#define KOUT    4096   // K*H

typedef __attribute__((ext_vector_type(16))) __bf16 v16bf;
typedef __attribute__((ext_vector_type(8)))  float  v8f;

union Frag { v16bf v; uint4 q[2]; };

__device__ __forceinline__ unsigned short f32_to_bf16_rne(float f) {
  unsigned int u = __float_as_uint(f);
  u += 0x7FFFu + ((u >> 16) & 1u);   // round-to-nearest-even
  return (unsigned short)(u >> 16);
}

// ---------------------------------------------------------------------------
// fp32 -> bf16 conversion, vectorized x4 (all sizes are multiples of 4)
// ---------------------------------------------------------------------------
__global__ void cvt_bf16_kernel(const float* __restrict__ src,
                                unsigned short* __restrict__ dst, int n4) {
  int i = blockIdx.x * blockDim.x + threadIdx.x;
  int stride = gridDim.x * blockDim.x;
  for (; i < n4; i += stride) {
    float4 f = ((const float4*)src)[i];
    ushort4 o;
    o.x = f32_to_bf16_rne(f.x);
    o.y = f32_to_bf16_rne(f.y);
    o.z = f32_to_bf16_rne(f.z);
    o.w = f32_to_bf16_rne(f.w);
    ((ushort4*)dst)[i] = o;
  }
}

// ---------------------------------------------------------------------------
// WMMA fragment loads (row-major sources, bf16 elements)
// A: 16x32 (MxK). Lane half h, row m: K = k0 + h*8 + [0..7] and k0+16+h*8+[0..7]
// B: 32x16 (KxN) fed from row-major W[N,K] (computing X @ W^T):
//    lane column n: K = k0 + half*8 + [0..7] and k0+16+half*8+[0..7]
// ---------------------------------------------------------------------------
__device__ __forceinline__ Frag load_a(const unsigned short* __restrict__ base,
                                       int ld, int row, int half, int k0) {
  Frag a;
  const unsigned short* p = base + (size_t)row * ld + k0 + half * 8;
  a.q[0] = *(const uint4*)p;
  a.q[1] = *(const uint4*)(p + 16);
  return a;
}

__device__ __forceinline__ Frag load_b(const unsigned short* __restrict__ base,
                                       int ld, int row, int half, int k0) {
  Frag b;
  const unsigned short* p = base + (size_t)row * ld + k0 + half * 8;
  b.q[0] = *(const uint4*)p;
  b.q[1] = *(const uint4*)(p + 16);
  return b;
}

__device__ __forceinline__ float selu_f(float x) {
  const float lam   = 1.0507009873554805f;
  const float alpha = 1.6732632423543772f;
  return x > 0.0f ? lam * x : lam * alpha * (__expf(x) - 1.0f);
}

// ---------------------------------------------------------------------------
// Hidden-state step: h_t = selu(X_t @ Wxh^T + b + h_cat @ Whh^T)  [B=32, H=1024]
// Each wave: both M-tiles x 1 N-tile (shares the B fragment across 2 WMMAs).
// 64 waves -> 8 blocks x 8 waves.
// ---------------------------------------------------------------------------
__global__ void __launch_bounds__(256)
hidden_step_kernel(const unsigned short* __restrict__ xbf_t,  // [B, IN] (this t)
                   const unsigned short* __restrict__ wxh,    // [H, IN] bf16
                   const unsigned short* __restrict__ whh,    // [H, KCAT] bf16
                   const float* __restrict__ bias,            // [H] f32
                   unsigned short* __restrict__ hbf,          // [T, B, H] bf16
                   int t) {
  int wave  = (blockIdx.x * blockDim.x + threadIdx.x) >> 5;
  int lane  = threadIdx.x & 31;
  int half  = lane >> 4;
  int l16   = lane & 15;
  int ntile = wave;                    // 0..63
  int n0 = ntile * 16;

  v8f acc0 = {};                       // rows 0..15
  v8f acc1 = {};                       // rows 16..31

  // Input projection: K = IN over X_t and Wxh
  for (int k0 = 0; k0 < IN_DIM; k0 += 32) {
    Frag a0 = load_a(xbf_t, IN_DIM, l16,      half, k0);
    Frag a1 = load_a(xbf_t, IN_DIM, 16 + l16, half, k0);
    Frag b  = load_b(wxh,   IN_DIM, n0 + l16, half, k0);
    acc0 = __builtin_amdgcn_wmma_f32_16x16x32_bf16(false, a0.v, false, b.v,
                                                   (short)0, acc0, false, false);
    acc1 = __builtin_amdgcn_wmma_f32_16x16x32_bf16(false, a1.v, false, b.v,
                                                   (short)0, acc1, false, false);
  }
  // Recurrent term: K = KCAT over [h_{t-1}, h_{t-2}, h_{t-3}] and Whh
  for (int k0 = 0; k0 < KCAT; k0 += 32) {
    int seg   = k0 >> 10;              // history block (wave-uniform branch)
    int src_t = t - 1 - seg;
    if (src_t < 0) continue;           // zero history contributes nothing
    const unsigned short* hsrc = hbf + (size_t)src_t * (B_DIM * H_DIM);
    int kl = k0 & (H_DIM - 1);
    Frag a0 = load_a(hsrc, H_DIM, l16,      half, kl);
    Frag a1 = load_a(hsrc, H_DIM, 16 + l16, half, kl);
    Frag b  = load_b(whh,  KCAT,  n0 + l16, half, k0);
    acc0 = __builtin_amdgcn_wmma_f32_16x16x32_bf16(false, a0.v, false, b.v,
                                                   (short)0, acc0, false, false);
    acc1 = __builtin_amdgcn_wmma_f32_16x16x32_bf16(false, a1.v, false, b.v,
                                                   (short)0, acc1, false, false);
  }

  // Bias + SELU, store h_t as bf16
  int n = n0 + l16;
  float bv = bias[n];
  unsigned short* hdst = hbf + (size_t)t * (B_DIM * H_DIM);
  #pragma unroll
  for (int j = 0; j < 8; ++j) {
    int m = j + half * 8;
    hdst[(size_t)m * H_DIM + n]        = f32_to_bf16_rne(selu_f(acc0[j] + bv));
    hdst[(size_t)(16 + m) * H_DIM + n] = f32_to_bf16_rne(selu_f(acc1[j] + bv));
  }
}

// ---------------------------------------------------------------------------
// Output projection: out_t = [h_t, h_cat] @ Wmo^T   [B=32, OUT=8192], K=4096
// Each wave: 2x2 tile quad (both M-tiles x 2 N-tiles): per K-step
// 2 A-frags + 2 B-frags -> 4 WMMAs (1KB loaded per WMMA).
// 256 waves -> 32 blocks x 8 waves.
// ---------------------------------------------------------------------------
__global__ void __launch_bounds__(256)
out_step_kernel(const unsigned short* __restrict__ hbf,  // [T, B, H] bf16
                const unsigned short* __restrict__ wmo,  // [OUT, KOUT] bf16
                float* __restrict__ out_t,               // [B, OUT] f32 slice
                int t) {
  int wave = (blockIdx.x * blockDim.x + threadIdx.x) >> 5;
  int lane = threadIdx.x & 31;
  int half = lane >> 4;
  int l16  = lane & 15;
  int n0   = wave * 32;                // 2 N-tiles: n0, n0+16   (wave 0..255)

  v8f acc00 = {};  // rows 0..15,  cols n0..n0+15
  v8f acc01 = {};  // rows 0..15,  cols n0+16..n0+31
  v8f acc10 = {};  // rows 16..31, cols n0..n0+15
  v8f acc11 = {};  // rows 16..31, cols n0+16..n0+31

  for (int k0 = 0; k0 < KOUT; k0 += 32) {
    int seg   = k0 >> 10;              // 0 -> h_t, 1 -> h_{t-1}, ...
    int src_t = t - seg;
    if (src_t < 0) continue;
    const unsigned short* hsrc = hbf + (size_t)src_t * (B_DIM * H_DIM);
    int kl = k0 & (H_DIM - 1);
    Frag a0 = load_a(hsrc, H_DIM, l16,           half, kl);
    Frag a1 = load_a(hsrc, H_DIM, 16 + l16,      half, kl);
    Frag b0 = load_b(wmo,  KOUT,  n0 + l16,      half, k0);
    Frag b1 = load_b(wmo,  KOUT,  n0 + 16 + l16, half, k0);
    acc00 = __builtin_amdgcn_wmma_f32_16x16x32_bf16(false, a0.v, false, b0.v,
                                                    (short)0, acc00, false, false);
    acc01 = __builtin_amdgcn_wmma_f32_16x16x32_bf16(false, a0.v, false, b1.v,
                                                    (short)0, acc01, false, false);
    acc10 = __builtin_amdgcn_wmma_f32_16x16x32_bf16(false, a1.v, false, b0.v,
                                                    (short)0, acc10, false, false);
    acc11 = __builtin_amdgcn_wmma_f32_16x16x32_bf16(false, a1.v, false, b1.v,
                                                    (short)0, acc11, false, false);
  }

  int n = n0 + l16;
  #pragma unroll
  for (int j = 0; j < 8; ++j) {
    int m = j + half * 8;
    out_t[(size_t)m * OUT_DIM + n]             = acc00[j];
    out_t[(size_t)m * OUT_DIM + n + 16]        = acc01[j];
    out_t[(size_t)(16 + m) * OUT_DIM + n]      = acc10[j];
    out_t[(size_t)(16 + m) * OUT_DIM + n + 16] = acc11[j];
  }
}

// ---------------------------------------------------------------------------
// Launch: convert weights/X to bf16 once, then 128 sequential (hidden, out)
// kernel pairs on `stream` (stream order provides the recurrence barrier).
// ---------------------------------------------------------------------------
extern "C" void kernel_launch(void* const* d_in, const int* in_sizes, int n_in,
                              void* d_out, int out_size, void* d_ws, size_t ws_size,
                              hipStream_t stream) {
  const float* X     = (const float*)d_in[0];  // [T, B, IN]
  const float* Wxh_w = (const float*)d_in[1];  // [H, IN]
  const float* Wxh_b = (const float*)d_in[2];  // [H]
  const float* Whh_w = (const float*)d_in[3];  // [H, (K-1)H]
  const float* Wmo_w = (const float*)d_in[4];  // [OUT, K*H]
  float* out = (float*)d_out;                  // [T, B, OUT]

  unsigned short* ws = (unsigned short*)d_ws;
  const size_t NX   = (size_t)T_STEPS * B_DIM * IN_DIM;   // 2,097,152
  const size_t NWXH = (size_t)H_DIM * IN_DIM;             //   524,288
  const size_t NWHH = (size_t)H_DIM * KCAT;               // 3,145,728
  const size_t NWMO = (size_t)OUT_DIM * KOUT;             // 33,554,432
  const size_t XBF = 0;
  const size_t WXH = XBF + NX;
  const size_t WHH = WXH + NWXH;
  const size_t WMO = WHH + NWHH;
  const size_t HBF = WMO + NWMO;                          // [T, B, H] bf16

  dim3 cb(256);
  cvt_bf16_kernel<<<dim3(1024), cb, 0, stream>>>(X,     ws + XBF, (int)(NX   / 4));
  cvt_bf16_kernel<<<dim3(512),  cb, 0, stream>>>(Wxh_w, ws + WXH, (int)(NWXH / 4));
  cvt_bf16_kernel<<<dim3(1024), cb, 0, stream>>>(Whh_w, ws + WHH, (int)(NWHH / 4));
  cvt_bf16_kernel<<<dim3(4096), cb, 0, stream>>>(Wmo_w, ws + WMO, (int)(NWMO / 4));

  for (int t = 0; t < T_STEPS; ++t) {
    hidden_step_kernel<<<dim3(8), dim3(256), 0, stream>>>(
        ws + XBF + (size_t)t * B_DIM * IN_DIM,
        ws + WXH, ws + WHH, Wxh_b, ws + HBF, t);
    out_step_kernel<<<dim3(32), dim3(256), 0, stream>>>(
        ws + HBF, ws + WMO, out + (size_t)t * B_DIM * OUT_DIM, t);
  }
}
